// NNModel2_4526895530075
// MI455X (gfx1250) — matrix operand order repaired and checked by reading, not compiled, not run on hardware
//
#include <hip/hip_runtime.h>

// ---------------------------------------------------------------------------
// FEM + MLP constitutive model for MI455X (gfx1250, wave32, WMMA).
//
// Per 16-point tile (one wave32):
//   phase A : gather u_e, B (kept in regs), strain, invariants  (VALU)
//   layer 1 : 2->64 tanh, h1 -> per-wave LDS as f16             (VALU)
//   layer 2 : 64->64 via 8x v_wmma_f32_16x16x32_f16             (matrix core)
//             W2 fragments re-read from LDS per n-tile to keep
//             VGPR pressure < 256 (no s_set_vgpr_msb juggling)
//   layer 3 : fused per n-tile: tanh + partials, shfl butterfly (VALU)
//   phase C : stress, E_P = B^T sigma w, 4-GP shfl reduction,
//             8 global_atomic_add_f32 per element
// ---------------------------------------------------------------------------

typedef __attribute__((ext_vector_type(16))) _Float16 v16h;
typedef __attribute__((ext_vector_type(8)))  _Float16 v8h;
typedef __attribute__((ext_vector_type(8)))  float    v8f;

#define SQ23F 0.81649658092772603273f

__device__ __forceinline__ float fast_tanh(float x) {
#if __has_builtin(__builtin_amdgcn_tanhf)
    return __builtin_amdgcn_tanhf(x);   // V_TANH_F32 (gfx1250 trans op)
#else
    return tanhf(x);
#endif
}

__global__ __launch_bounds__(256) void fem_mlp_wmma(
    const float* __restrict__ u,            // [NDOF]
    const float* __restrict__ B,            // [E,4,3,8]
    const float* __restrict__ Jacc,         // [E,4]
    const float* __restrict__ gp_w,         // [4]
    const int*   __restrict__ conn,         // [E,4]
    const float* __restrict__ weight1,      // [NDOF]
    const float* __restrict__ scales_inp,   // [2]
    const float* __restrict__ limits_inp,   // [2]
    const float* __restrict__ scales_grad,  // [2]
    const float* __restrict__ limits_grad,  // [2]
    const float* __restrict__ W1,           // [2,64]
    const float* __restrict__ b1,           // [64]
    const float* __restrict__ W2,           // [64,64]
    const float* __restrict__ b2,           // [64]
    const float* __restrict__ W3,           // [64,2]
    const float* __restrict__ b3,           // [2]
    float*       __restrict__ F,            // [NDOF] (pre-zeroed)
    int nelem)
{
    // LDS: W2 transposed f16 (8KB) + W1/b1 (0.75KB) + per-wave h1 (16KB) + g (1KB)
    __shared__ _Float16 sW2h[64 * 64];      // sW2h[n*64 + k] = (f16)W2[k][n]
    __shared__ float    sW1[2 * 64];
    __shared__ float    sb1[64];
    __shared__ _Float16 sH1[8][16 * 64];    // per-wave h1 tile (row-major [M][K])
    __shared__ float    sG[8][16 * 2];      // per-wave layer-3 output [M][2]

    const int tid = threadIdx.x;

    // ---- block-wide weight staging ----
    for (int i = tid; i < 64 * 64; i += 256) {
        const int k = i >> 6, n = i & 63;
        sW2h[n * 64 + k] = (_Float16)W2[i];
    }
    if (tid < 128) sW1[tid] = W1[tid];
    if (tid < 64)  sb1[tid] = b1[tid];
    __syncthreads();

    const int lane = tid & 31;
    const int wib  = tid >> 5;              // wave-in-block (0..7)
    const int lm   = lane & 15;             // point-in-tile / WMMA N-index / A row M
    const int hi   = lane >> 4;             // half select (0/1)

    // ---- per-lane loop-invariant preloads ----
    float b2l[4], w3a[4], w3b[4];
#pragma unroll
    for (int nt = 0; nt < 4; ++nt) {
        const int n = nt * 16 + lm;
        b2l[nt] = b2[n];
        w3a[nt] = W3[n * 2 + 0];
        w3b[nt] = W3[n * 2 + 1];
    }
    const float si0 = scales_inp[0],  si1 = scales_inp[1];
    const float li0 = limits_inp[0],  li1 = limits_inp[1];
    const float sg0 = scales_grad[0], sg1 = scales_grad[1];
    const float lg0 = limits_grad[0], lg1 = limits_grad[1];
    const float b30 = b3[0], b31 = b3[1];
    const float gpw = gp_w[lane & 3];       // gauss-point id == lane&3

    // Base half-index of this lane's W2 B-fragment rows in sW2h:
    // B 32x16 fragments as loaded here (same k order as A): N = lane%16; lanes 0-15 hold K 0..15,
    // lanes 16-31 hold K 16..31 (16 contiguous halves per lane).
    const int wbase = lm * 64 + hi * 16;    // + nt*16*64 + chunk*32 at use site

    const int npoints = nelem * 4;
    const int ntiles  = (npoints + 15) >> 4;
    const int gwave   = (blockIdx.x << 3) + wib;
    const int nwaves  = (int)(gridDim.x << 3);

    for (int tile = gwave; tile < ntiles; tile += nwaves) {
        // opaque zero: keeps per-nt W2 fragment loads inside the loop so the
        // register allocator never keeps all 64 W2 VGPRs resident at once
        int zv = 0;
        asm volatile("" : "+v"(zv));

        const int  p     = tile * 16 + lm;          // global gauss point
        const bool valid = (p < npoints);
        const int  pc    = valid ? p : 0;
        const int  e     = pc >> 2;

        // ================= Phase A: gather + strain + invariants =========
        int edof[8];
#pragma unroll
        for (int k = 0; k < 4; ++k) {
            const int n = conn[e * 4 + k];
            edof[2 * k]     = 2 * n;
            edof[2 * k + 1] = 2 * n + 1;
        }
        float ue[8];
#pragma unroll
        for (int j = 0; j < 8; ++j)
            ue[j] = weight1[edof[j]] * u[edof[j]];

        float Breg[24];                              // B[e][gp], kept for phase C
        const float* bp = B + (size_t)pc * 24;
#pragma unroll
        for (int j = 0; j < 6; ++j) {
            const float4 t = ((const float4*)bp)[j];
            Breg[4 * j + 0] = t.x; Breg[4 * j + 1] = t.y;
            Breg[4 * j + 2] = t.z; Breg[4 * j + 3] = t.w;
        }
        float s0 = 0.f, s1 = 0.f, s2 = 0.f;
#pragma unroll
        for (int j = 0; j < 8; ++j) {
            s0 += Breg[j]      * ue[j];
            s1 += Breg[8 + j]  * ue[j];
            s2 += Breg[16 + j] * ue[j];
        }
        const float ev  = s0 + s1;
        const float d00 = s0 - ev * (1.f / 3.f);
        const float d11 = s1 - ev * (1.f / 3.f);
        const float d22 = -ev * (1.f / 3.f);
        const float d01 = 0.5f * s2;
        const float det = sqrtf(d00 * d00 + d11 * d11 + d22 * d22 + 2.f * d01 * d01);
        const float rdet = 1.f / det;
        const float n00 = d00 * rdet, n11 = d11 * rdet, n01 = d01 * rdet;
        const float es  = det * SQ23F;
        const float ns0 = valid ? (ev * si0 + li0) : 0.f;
        const float ns1 = valid ? (es * si1 + li1) : 0.f;
        const float wsc = Jacc[pc] * gpw;            // WIDTH == 1.0

        // ================= Layer 1 (2->64, tanh) -> LDS f16 ==============
        _Float16* h1row = &sH1[wib][lm * 64];        // this lane's point row
#pragma unroll
        for (int jb = 0; jb < 4; ++jb) {
            v8h pack;
#pragma unroll
            for (int jj = 0; jj < 8; ++jj) {
                const int j = hi * 32 + jb * 8 + jj;
                const float z = ns0 * sW1[j] + ns1 * sW1[64 + j] + sb1[j];
                pack[jj] = (_Float16)fast_tanh(z);
            }
            *(v8h*)&h1row[hi * 32 + jb * 8] = pack;  // 16B aligned ds_store
        }
        // same-wave DS ops are in-order: stores above complete before loads below

        // ====== Layer 2 (WMMA) fused with layer-3 partial dot products ===
        // 16-bit A 16x32 layout: row M = lane%16; lane<16 -> K c*32+0..15;
        // lane>=16 -> K c*32+16..31 (contiguous, matching B fragments).
        union { v16h v; v8h h[2]; } afr[2];
#pragma unroll
        for (int c = 0; c < 2; ++c) {
            const int kb = c * 32 + hi * 16;
            afr[c].h[0] = *(const v8h*)&h1row[kb];
            afr[c].h[1] = *(const v8h*)&h1row[kb + 8];
        }
        float pg[8][2];
#pragma unroll
        for (int r = 0; r < 8; ++r) { pg[r][0] = 0.f; pg[r][1] = 0.f; }

#pragma unroll
        for (int nt = 0; nt < 4; ++nt) {
            // re-read this n-tile's W2 fragments from LDS (zv defeats hoisting)
            const v16h w0 = *(const v16h*)&sW2h[nt * (16 * 64) + wbase + zv];
            const v16h w1 = *(const v16h*)&sW2h[nt * (16 * 64) + wbase + 32 + zv];
            v8f z = {};
            z = __builtin_amdgcn_wmma_f32_16x16x32_f16(
                    false, afr[0].v, false, w0, (short)0, z, false, false);
            z = __builtin_amdgcn_wmma_f32_16x16x32_f16(
                    false, afr[1].v, false, w1, (short)0, z, false, false);
            // C/D layout: VGPR r holds (M = r + hi*8, N = nt*16 + lane%16)
#pragma unroll
            for (int r = 0; r < 8; ++r) {
                const float h2 = fast_tanh(z[r] + b2l[nt]);
                pg[r][0] += h2 * w3a[nt];
                pg[r][1] += h2 * w3b[nt];
            }
        }

        // reduce over N (16 lanes of each half, xor masks 1,2,4,8)
#pragma unroll
        for (int m = 1; m <= 8; m <<= 1)
#pragma unroll
            for (int r = 0; r < 8; ++r) {
                pg[r][0] += __shfl_xor(pg[r][0], m, 32);
                pg[r][1] += __shfl_xor(pg[r][1], m, 32);
            }
        // publish rows: lane0 -> M 0..7, lane16 -> M 8..15
        float* g = sG[wib];
        if (lane == 0) {
#pragma unroll
            for (int r = 0; r < 8; ++r) {
                g[2 * r]     = pg[r][0] + b30;
                g[2 * r + 1] = pg[r][1] + b31;
            }
        }
        if (lane == 16) {
#pragma unroll
            for (int r = 0; r < 8; ++r) {
                g[2 * (8 + r)]     = pg[r][0] + b30;
                g[2 * (8 + r) + 1] = pg[r][1] + b31;
            }
        }
        const float g0 = g[2 * lm];
        const float g1 = g[2 * lm + 1];

        // ================= Phase C: stress, E_P, scatter ================
        const float pinv = (g0 - lg0) / sg0;
        const float qinv = (g1 - lg1) / sg1;
        const float cc   = SQ23F * qinv;
        const float t0 = (pinv + cc * n00) * wsc;
        const float t1 = (pinv + cc * n11) * wsc;
        const float t2 = (cc * n01) * wsc;
        float f[8];
#pragma unroll
        for (int j = 0; j < 8; ++j) {
            float v = Breg[j] * t0 + Breg[8 + j] * t1 + Breg[16 + j] * t2;
            f[j] = valid ? v : 0.f;
        }
        // sum the 4 gauss points of this element (lanes L^1, L^2)
#pragma unroll
        for (int j = 0; j < 8; ++j) {
            f[j] += __shfl_xor(f[j], 1, 32);
            f[j] += __shfl_xor(f[j], 2, 32);
        }
        if (((lane & 3) == 0) && lane < 16 && valid) {
#pragma unroll
            for (int j = 0; j < 8; ++j)
                atomicAdd(&F[edof[j]], f[j]);
        }
    }
}

extern "C" void kernel_launch(void* const* d_in, const int* in_sizes, int n_in,
                              void* d_out, int out_size, void* d_ws, size_t ws_size,
                              hipStream_t stream) {
    const float* u            = (const float*)d_in[0];
    const float* B            = (const float*)d_in[1];
    const float* Jacc         = (const float*)d_in[2];
    const float* gp_w         = (const float*)d_in[3];
    const int*   conn         = (const int*)  d_in[4];
    const float* weight1      = (const float*)d_in[5];
    const float* scales_inp   = (const float*)d_in[6];
    const float* limits_inp   = (const float*)d_in[7];
    const float* scales_grad  = (const float*)d_in[8];
    const float* limits_grad  = (const float*)d_in[9];
    const float* W1           = (const float*)d_in[10];
    const float* b1           = (const float*)d_in[11];
    const float* W2           = (const float*)d_in[12];
    const float* b2           = (const float*)d_in[13];
    const float* W3           = (const float*)d_in[14];
    const float* b3           = (const float*)d_in[15];
    float*       F            = (float*)d_out;

    const int nelem = in_sizes[2] / 4;   // Jacc is [E,4]

    // output is assembled with atomics -> must start from zero
    hipMemsetAsync(d_out, 0, (size_t)out_size * sizeof(float), stream);

    const int blocks = 2048;             // 16384 waves, grid-stride over 125k tiles
    fem_mlp_wmma<<<blocks, 256, 0, stream>>>(
        u, B, Jacc, gp_w, conn, weight1, scales_inp, limits_inp,
        scales_grad, limits_grad, W1, b1, W2, b2, W3, b3, F, nelem);
}
